// PhotoMetric_446676598882
// MI455X (gfx1250) — compile-verified, matrix operand order checked by reading, not run on hardware
//
#include <hip/hip_runtime.h>
#include <math.h>

typedef __attribute__((ext_vector_type(2)))  float    v2f;
typedef __attribute__((ext_vector_type(8)))  float    v8f;
typedef __attribute__((ext_vector_type(16))) _Float16 v16h;

#define BATCH 16
#define HH    512
#define WW    512
#define PLANE (HH * WW)            // 262144 = 2^18
#define NPIX  (BATCH * PLANE)      // 4194304
#define NBLK  2048
#define NTHR  256                  // 8 waves of 32

// ---- wave32 sum via WMMA: A holds per-lane partials (K=0 / K=2 slots),
// ---- B = ones => D[m][n] = acc(m) + acc(m+16). Sum the 8 C VGPRs, then one
// ---- shfl_xor(16) combines the two half-wave row groups.
__device__ __forceinline__ float wave_reduce_wmma(float acc) {
    v8f c = {0.f, 0.f, 0.f, 0.f, 0.f, 0.f, 0.f, 0.f};
#if __has_builtin(__builtin_amdgcn_wmma_f32_16x16x4_f32)
    v2f a; a.x = acc;  a.y = 0.0f;
    v2f b; b.x = 1.0f; b.y = 1.0f;
    c = __builtin_amdgcn_wmma_f32_16x16x4_f32(false, a, false, b,
                                              (short)0, c, false, false);
#else
    // fallback: codegen-confirmed f16 WMMA; same A/B trick (K=0 and K=8 slots)
    v16h a = {};
    a[0] = (_Float16)acc;
    v16h b;
    #pragma unroll
    for (int i = 0; i < 16; ++i) b[i] = (_Float16)1.0f;
    c = __builtin_amdgcn_wmma_f32_16x16x32_f16(false, a, false, b,
                                               (short)0, c, false, false);
#endif
    float t = (c[0] + c[1]) + (c[2] + c[3]) + (c[4] + c[5]) + (c[6] + c[7]);
    t += __shfl_xor(t, 16, 32);   // full wave32 sum in every lane
    return t;
}

__device__ __forceinline__ float charbonnier(float d) {
    float t = __builtin_fmaf(d, d, 1.0e-6f);           // p^2 + DELTA^2, >= 1e-6
#if __has_builtin(__builtin_amdgcn_logf) && __has_builtin(__builtin_amdgcn_exp2f)
    return __builtin_amdgcn_exp2f(0.45f * __builtin_amdgcn_logf(t));
#else
    return exp2f(0.45f * log2f(t));                    // host-safe fallback
#endif
}

__global__ void __launch_bounds__(NTHR)
photo_partial(const float* __restrict__ f1, const float* __restrict__ f2,
              const float* __restrict__ fl, float* __restrict__ partial) {
    const int tid = blockIdx.x * NTHR + threadIdx.x;   // 524288 threads total
    float acc = 0.0f;

    #pragma unroll 2
    for (int base = 0; base < NPIX; base += NBLK * NTHR) {
        const int pix = base + tid;                    // always < NPIX (exact division)
        const int b   = pix >> 18;
        const int rem = pix & (PLANE - 1);
        const int y   = rem >> 9;
        const int x   = rem & (WW - 1);

        const float fx = fl[(b * 2 + 0) * PLANE + rem];
        const float fy = fl[(b * 2 + 1) * PLANE + rem];

        // grid_sample(align_corners=True, border), grid normalized by W (not W-1)
        float ix = ((float)x + fx) * (511.0f / 512.0f);
        float iy = ((float)y + fy) * (511.0f / 512.0f);
        ix = fminf(fmaxf(ix, 0.0f), 511.0f);
        iy = fminf(fmaxf(iy, 0.0f), 511.0f);

        const float x0f = floorf(ix), y0f = floorf(iy);
        const float wx = ix - x0f,    wy = iy - y0f;
        const int x0 = (int)x0f,      y0 = (int)y0f;
        const int x1 = min(x0 + 1, WW - 1);
        const int y1 = min(y0 + 1, HH - 1);

        const int i00 = y0 * WW + x0, i01 = y0 * WW + x1;
        const int i10 = y1 * WW + x0, i11 = y1 * WW + x1;

        const float* __restrict__ p2 = f2 + (size_t)b * 3 * PLANE;
        const float* __restrict__ p1 = f1 + (size_t)b * 3 * PLANE + rem;

        const float w00 = (1.0f - wx) * (1.0f - wy);
        const float w01 = wx * (1.0f - wy);
        const float w10 = (1.0f - wx) * wy;
        const float w11 = wx * wy;

        #pragma unroll
        for (int ch = 0; ch < 3; ++ch) {
            const float* __restrict__ pc = p2 + ch * PLANE;
            float v = pc[i00] * w00 + pc[i01] * w01 + pc[i10] * w10 + pc[i11] * w11;
            acc += charbonnier(p1[ch * PLANE] - v);
        }
    }

    // wave32 reduce (v_wmma), then cross-wave via LDS; one partial per block.
    const float wsum = wave_reduce_wmma(acc);
    __shared__ float lds[NTHR / 32];
    const int lane = threadIdx.x & 31;
    const int wid  = threadIdx.x >> 5;
    if (lane == 0) lds[wid] = wsum;
    __syncthreads();
    if (threadIdx.x == 0) {
        float s = 0.0f;
        #pragma unroll
        for (int k = 0; k < NTHR / 32; ++k) s += lds[k];
        partial[blockIdx.x] = s;
    }
}

__global__ void __launch_bounds__(NTHR)
reduce_final(const float* __restrict__ partial, float* __restrict__ out) {
    float acc = 0.0f;
    #pragma unroll
    for (int i = 0; i < NBLK; i += NTHR)               // 8 uniform iterations
        acc += partial[i + threadIdx.x];

    const float wsum = wave_reduce_wmma(acc);
    __shared__ float lds[NTHR / 32];
    const int lane = threadIdx.x & 31;
    const int wid  = threadIdx.x >> 5;
    if (lane == 0) lds[wid] = wsum;
    __syncthreads();
    if (threadIdx.x == 0) {
        float s = 0.0f;
        #pragma unroll
        for (int k = 0; k < NTHR / 32; ++k) s += lds[k];
        out[0] = s;
    }
}

extern "C" void kernel_launch(void* const* d_in, const int* in_sizes, int n_in,
                              void* d_out, int out_size, void* d_ws, size_t ws_size,
                              hipStream_t stream) {
    const float* frame1 = (const float*)d_in[0];   // (16,3,512,512) f32
    const float* frame2 = (const float*)d_in[1];   // (16,3,512,512) f32
    const float* flow   = (const float*)d_in[2];   // (16,2,512,512) f32
    float*       out    = (float*)d_out;           // scalar f32
    float*       part   = (float*)d_ws;            // 2048 floats (8 KB) scratch

    photo_partial<<<NBLK, NTHR, 0, stream>>>(frame1, frame2, flow, part);
    reduce_final<<<1, NTHR, 0, stream>>>(part, out);
}
